// IITTransformerLayer_58909771432593
// MI455X (gfx1250) — matrix operand order repaired and checked by reading, not compiled, not run on hardware
//
#include <hip/hip_runtime.h>

// ---------------------------------------------------------------------------
// IIT transformer layer for MI455X (gfx1250, wave32, WMMA + async/TDM staging).
// GEMMs: f32 global --(GLOBAL_LOAD_ASYNC_TO_LDS / TENSOR_LOAD_TO_LDS)--> LDS
// f32 stage --> bf16 LDS tiles --> v_wmma_f32_16x16x32_bf16 --> f32 out.
// ---------------------------------------------------------------------------

#define TPB 256

typedef __attribute__((ext_vector_type(16))) __bf16 v16bf;
typedef __attribute__((ext_vector_type(8)))  float  v8f;
typedef __attribute__((ext_vector_type(4)))  unsigned int u32x4;
typedef __attribute__((ext_vector_type(8)))  int  i32x8;
typedef __attribute__((ext_vector_type(4)))  int  i32x4;

__device__ __forceinline__ unsigned short f32_to_bf16(float f) {
  union { float f; unsigned u; } cv; cv.f = f;
  unsigned r = cv.u + 0x7FFFu + ((cv.u >> 16) & 1u);   // round-to-nearest-even
  return (unsigned short)(r >> 16);
}
#define BF(x) __builtin_bit_cast(__bf16, (unsigned short)(x))

__device__ __forceinline__ float geluf(float x) {
  return 0.5f * x * (1.0f + erff(x * 0.70710678118654752440f));
}

// LDS byte offset of a __shared__ object (low 32 bits of the flat address).
__device__ __forceinline__ unsigned lds_byte_off(const void* p) {
  return (unsigned)(unsigned long long)p;
}

// GLOBAL_LOAD_ASYNC_TO_LDS_B128, GVS mode (SGPR base + VGPR 32-bit offset).
__device__ __forceinline__ void async_load_b128(unsigned ldsAddr, unsigned gOff,
                                                unsigned long long base) {
  asm volatile("global_load_async_to_lds_b128 %0, %1, %2"
               :: "v"(ldsAddr), "v"(gOff), "s"(base) : "memory");
}
__device__ __forceinline__ void wait_async0() {
  asm volatile("s_wait_asynccnt 0" ::: "memory");
}

// ---------------------------------------------------------------------------
// TDM: load a rows x cols f32 tile (row stride = rowStride elems) into LDS.
// 2-D descriptor per ISA 8.3/8.4: count=1, type=2, data_size=4B.
// ---------------------------------------------------------------------------
__device__ __forceinline__ void tdm_load_tile_f32(const float* gsrc, unsigned ldsOff,
                                                  int rowStride, int rows, int cols,
                                                  int tensorRows) {
  const unsigned long long ga = (unsigned long long)gsrc;
  u32x4 g0;
  g0[0] = 1u;                                                 // count=1, user mode
  g0[1] = ldsOff;                                             // lds_addr [63:32]
  g0[2] = (unsigned)(ga & 0xFFFFFFFFull);                     // global_addr lo
  g0[3] = (unsigned)((ga >> 32) & 0x1FFFFFFull) | (2u << 30); // addr hi | type=2
  i32x8 g1;
  g1[0] = (int)(2u << 16);                                    // data_size = 4B
  g1[1] = (int)(((unsigned)rowStride & 0xFFFFu) << 16);       // tensor_dim0 lo16
  g1[2] = (int)((((unsigned)rowStride >> 16) & 0xFFFFu) |
                (((unsigned)tensorRows & 0xFFFFu) << 16));    // dim0 hi | dim1 lo
  g1[3] = (int)((((unsigned)tensorRows >> 16) & 0xFFFFu) |
                (((unsigned)cols & 0xFFFFu) << 16));          // dim1 hi | tile_dim0
  g1[4] = (int)((unsigned)rows & 0xFFFFu);                    // tile_dim1
  g1[5] = (int)rowStride;                                     // tensor_dim0_stride
  g1[6] = 0;
  g1[7] = 0;
  i32x4 z4; z4[0] = 0; z4[1] = 0; z4[2] = 0; z4[3] = 0;       // 2-D: groups 2/3 unused
#if defined(__clang_major__) && (__clang_major__ >= 23)
  i32x8 z8; for (int i = 0; i < 8; ++i) z8[i] = 0;
  __builtin_amdgcn_tensor_load_to_lds(g0, g1, z4, z4, z8, 0);
#else
  __builtin_amdgcn_tensor_load_to_lds(g0, g1, z4, z4, 0);
#endif
}

// ---------------------------------------------------------------------------
// LayerNorm over last dim D (D in {256,512,1024}), one block per row.
// ACT==1 fuses exact GELU after LN. Supports in-place (in==out).
// ---------------------------------------------------------------------------
template <int ACT>
__global__ __launch_bounds__(TPB)
void ln_kernel(const float* in, const float* g, const float* b,
               float* out, int D) {
  const int row = blockIdx.x;
  const int tid = threadIdx.x;
  const float* r = in + (size_t)row * D;
  float* o = out + (size_t)row * D;
  const int per = D >> 8;                 // D / 256
  float vals[4];
  float s = 0.f;
  for (int i = 0; i < per; ++i) { vals[i] = r[tid + (i << 8)]; s += vals[i]; }
  __shared__ float red[TPB];
  red[tid] = s; __syncthreads();
  for (int off = 128; off > 0; off >>= 1) {
    if (tid < off) red[tid] += red[tid + off];
    __syncthreads();
  }
  const float mean = red[0] / (float)D;
  __syncthreads();
  float vs = 0.f;
  for (int i = 0; i < per; ++i) { float d = vals[i] - mean; vs += d * d; }
  red[tid] = vs; __syncthreads();
  for (int off = 128; off > 0; off >>= 1) {
    if (tid < off) red[tid] += red[tid + off];
    __syncthreads();
  }
  const float inv = rsqrtf(red[0] / (float)D + 1e-5f);
  for (int i = 0; i < per; ++i) {
    const int c = tid + (i << 8);
    float y = (vals[i] - mean) * inv * g[c] + b[c];
    if (ACT == 1) y = geluf(y);
    o[c] = y;
  }
}

// ---------------------------------------------------------------------------
// Gate head: out[row] = sigmoid(dot(in[row,:], w) + bias[0]); block per row.
// ---------------------------------------------------------------------------
__global__ __launch_bounds__(TPB)
void dot_sigmoid_kernel(const float* __restrict__ in, const float* __restrict__ w,
                        const float* __restrict__ bias, float* __restrict__ out,
                        int D) {
  const int row = blockIdx.x, tid = threadIdx.x;
  const float* r = in + (size_t)row * D;
  float s = 0.f;
  for (int c = tid; c < D; c += TPB) s += r[c] * w[c];
  __shared__ float red[TPB];
  red[tid] = s; __syncthreads();
  for (int off = 128; off > 0; off >>= 1) {
    if (tid < off) red[tid] += red[tid + off];
    __syncthreads();
  }
  if (tid == 0) out[row] = 1.f / (1.f + __expf(-(red[0] + bias[0])));
}

// ---------------------------------------------------------------------------
// phi_bias[b][h][t] = tanh(phi[b][t] * pb_w[h] + pb_b[h]);  (B,H,T) layout
// ---------------------------------------------------------------------------
__global__ __launch_bounds__(TPB)
void phibias_kernel(const float* __restrict__ phi, const float* __restrict__ pbw,
                    const float* __restrict__ pbb, float* __restrict__ out,
                    int B, int H, int T) {
  const int i = blockIdx.x * TPB + threadIdx.x;
  if (i < B * H * T) {
    const int t = i % T;
    const int h = (i / T) % H;
    const int b = i / (T * H);
    out[i] = tanhf(phi[b * T + t] * pbw[h] + pbb[h]);
  }
}

// ---------------------------------------------------------------------------
// out = a*xa + (1-a)*xb, gate per token (row of width D)
// ---------------------------------------------------------------------------
__global__ __launch_bounds__(TPB)
void blend_kernel(const float* __restrict__ gate, const float* __restrict__ xa,
                  const float* __restrict__ xb, float* __restrict__ out,
                  int D, long n) {
  const long i = (long)blockIdx.x * TPB + threadIdx.x;
  if (i < n) {
    const float a = gate[i / D];
    out[i] = a * xa[i] + (1.f - a) * xb[i];
  }
}

// ---------------------------------------------------------------------------
// Tiled bf16 WMMA GEMM:  C[M,N] = act( A[M,K] @ Bw[K,N] + bias )
// A may be split: columns [0,K1) from A1 (stride K1), [K1,K) from A2
// (stride K-K1) — used for concat(hidden, attn_out) @ W gates.
// Staging: A via GLOBAL_LOAD_ASYNC_TO_LDS_B128 (ASYNCcnt), B via TDM
// TENSOR_LOAD_TO_LDS (TENSORcnt, wave 0 only). 1-deep software pipeline:
// chunk k+1 DMA is issued before chunk k's WMMAs.
// Block tile 128x128, 8 waves of 32 lanes; wave tile 32x64 (2x4 of 16x16).
// ---------------------------------------------------------------------------
#define BK 32

template <int ACT>
__global__ __launch_bounds__(TPB)
void gemm_bf16_wmma(const float* __restrict__ A1, int K1,
                    const float* __restrict__ A2,
                    const float* __restrict__ Bw,
                    const float* __restrict__ bias,
                    float* __restrict__ C, int M, int N, int K) {
  __shared__ float stA[128][BK];               // f32 stage, filled by async loads
  __shared__ float stB[BK][128];               // f32 stage, filled by TDM
  __shared__ unsigned short lA[128][BK + 2];   // bf16 [m][k]
  __shared__ unsigned short lB[128][BK + 2];   // bf16 [n][k]

  const int tid  = threadIdx.x;
  const int lane = tid & 31;
  const int wave = tid >> 5;
  const int wm   = wave & 3;       // 0..3 -> 32-row stripe
  const int wn   = wave >> 2;      // 0..1 -> 64-col stripe
  const int hi   = lane >> 4;
  const int l15  = lane & 15;
  const int bm0  = blockIdx.y * 128;
  const int bn0  = blockIdx.x * 128;
  const int K2   = K - K1;

  v8f acc[2][4];
  for (int i = 0; i < 2; ++i)
    for (int j = 0; j < 4; ++j)
      for (int e = 0; e < 8; ++e) acc[i][j][e] = 0.f;

  // issue async A-tile loads + TDM B-tile load for chunk starting at k0
  auto issue_chunk = [&](int k0) {
    const float* Ab; int As; int kc;
    if (k0 < K1) { Ab = A1; As = K1; kc = k0; }        // BK divides K1: no straddle
    else         { Ab = A2; As = K2; kc = k0 - K1; }
    const unsigned long long abase = (unsigned long long)Ab;
    for (int i = 0; i < 4; ++i) {                      // 128x32 f32 = 1024 b128 ops
      const int idx = tid + TPB * i;
      const int ml = idx >> 3;
      const int kl = (idx & 7) << 2;
      const unsigned gOff = (unsigned)(((size_t)(bm0 + ml) * As + kc + kl) * 4u);
      async_load_b128(lds_byte_off(&stA[ml][kl]), gOff, abase);
    }
    if (wave == 0)
      tdm_load_tile_f32(Bw + (size_t)k0 * N + bn0, lds_byte_off(&stB[0][0]),
                        N, BK, 128, K);
  };

  issue_chunk(0);

  for (int k0 = 0; k0 < K; k0 += BK) {
    // wait for this chunk's DMA (per-wave counters), then block-sync
    __builtin_amdgcn_s_wait_tensorcnt(0);
    wait_async0();
    __syncthreads();

    // convert f32 stage -> bf16 compute tiles
    for (int i = 0; i < (128 * BK) / TPB; ++i) {
      const int idx = tid + TPB * i;
      const int ml = idx >> 5;
      const int kl = idx & (BK - 1);
      lA[ml][kl] = f32_to_bf16(stA[ml][kl]);
    }
    for (int i = 0; i < (128 * BK) / TPB; ++i) {
      const int idx = tid + TPB * i;
      const int nl = idx & 127;
      const int kl = idx >> 7;
      lB[nl][kl] = f32_to_bf16(stB[kl][nl]);
    }
    __syncthreads();

    // stage buffers are free: overlap next chunk's DMA with this chunk's WMMAs
    if (k0 + BK < K) issue_chunk(k0 + BK);

    // ---- fragments (contiguous k = 16*hi + s, same order in A and B) ------------------------------
    v16bf af[2], bf[4];
    for (int ti = 0; ti < 2; ++ti) {
      const int m = wm * 32 + ti * 16 + l15;
      for (int vv = 0; vv < 8; ++vv) {
        const int kk = 2 * vv + 16 * hi;
        af[ti][2 * vv]     = BF(lA[m][kk]);
        af[ti][2 * vv + 1] = BF(lA[m][kk + 1]);
      }
    }
    for (int tj = 0; tj < 4; ++tj) {
      const int n = wn * 64 + tj * 16 + l15;
      for (int vv = 0; vv < 8; ++vv) {
        const int kk = 2 * vv + 16 * hi;
        bf[tj][2 * vv]     = BF(lB[n][kk]);
        bf[tj][2 * vv + 1] = BF(lB[n][kk + 1]);
      }
    }
    for (int ti = 0; ti < 2; ++ti)
      for (int tj = 0; tj < 4; ++tj)
        acc[ti][tj] = __builtin_amdgcn_wmma_f32_16x16x32_bf16(
            false, af[ti], false, bf[tj], (short)0, acc[ti][tj], false, false);
    __syncthreads();   // lA/lB consumed; next convert may overwrite
  }

  // ---- epilogue: bias (+gelu) ----------------------------------------------
  for (int ti = 0; ti < 2; ++ti) {
    for (int tj = 0; tj < 4; ++tj) {
      const int colg = bn0 + wn * 64 + tj * 16 + l15;
      const float bv = bias ? bias[colg] : 0.f;
      for (int vv = 0; vv < 8; ++vv) {
        const int rowg = bm0 + wm * 32 + ti * 16 + hi * 8 + vv;
        float x = acc[ti][tj][vv] + bv;
        if (ACT == 1) x = geluf(x);
        C[(size_t)rowg * N + colg] = x;
      }
    }
  }
}

// ---------------------------------------------------------------------------
// Flash attention, per (b,h) and 128-query tile. HD=64, T=1024, H=16.
// Each of 8 waves owns 16 queries; K/V chunks of 64 keys staged in LDS.
// S = Q K^T and O += P V both via v_wmma_f32_16x16x32_bf16.
// phi bias (per query row, constant over keys) added before softmax.
// ---------------------------------------------------------------------------
__global__ __launch_bounds__(TPB)
void attn_flash_kernel(const float* __restrict__ Qg, const float* __restrict__ Kg,
                       const float* __restrict__ Vg, const float* __restrict__ pbias,
                       float* __restrict__ ctx) {
  const int T = 1024, H = 16, HD = 64, DM = 1024;
  __shared__ unsigned short sQ[128][HD + 2];      // [q][d]
  __shared__ unsigned short sK[64][HD + 2];       // [key][d]
  __shared__ unsigned short sV[HD][64 + 2];       // [d][key] (transposed)
  __shared__ unsigned short sP[8][16][64 + 2];    // per-wave P tile [q][key]

  const int tid  = threadIdx.x;
  const int lane = tid & 31;
  const int wave = tid >> 5;
  const int hi   = lane >> 4;
  const int l15  = lane & 15;
  const int bh   = blockIdx.y;
  const int b    = bh / H;
  const int h    = bh % H;
  const int q0   = blockIdx.x * 128;
  const float scale = 0.125f;  // 1/sqrt(64)

  // stage Q tile (scale folded into bf16 conversion)
  for (int i = 0; i < (128 * 64) / TPB; ++i) {
    const int idx = tid + TPB * i;
    const int d = idx & 63, qq = idx >> 6;
    sQ[qq][d] = f32_to_bf16(Qg[((size_t)b * T + q0 + qq) * DM + h * HD + d] * scale);
  }

  float qb[8], mrow[8], lsum[8];
  for (int r = 0; r < 8; ++r) {
    const int qq = q0 + wave * 16 + hi * 8 + r;
    qb[r]   = pbias[((size_t)b * H + h) * T + qq];
    mrow[r] = -3.0e38f;
    lsum[r] = 0.f;
  }
  v8f o[4];
  for (int j = 0; j < 4; ++j)
    for (int e = 0; e < 8; ++e) o[j][e] = 0.f;

  __syncthreads();

  // Q fragments are invariant across key chunks
  v16bf aq[2];
  {
    const int m = wave * 16 + l15;
    for (int c = 0; c < 2; ++c)
      for (int vv = 0; vv < 8; ++vv) {
        const int kk = c * 32 + 2 * vv + 16 * hi;
        aq[c][2 * vv]     = BF(sQ[m][kk]);
        aq[c][2 * vv + 1] = BF(sQ[m][kk + 1]);
      }
  }

  for (int kc = 0; kc < T; kc += 64) {
    // stage K chunk [key][d] and V chunk transposed [d][key]
    for (int i = 0; i < (64 * 64) / TPB; ++i) {
      const int idx = tid + TPB * i;
      const int d = idx & 63, kk = idx >> 6;
      sK[kk][d] = f32_to_bf16(Kg[((size_t)b * T + kc + kk) * DM + h * HD + d]);
      sV[d][kk] = f32_to_bf16(Vg[((size_t)b * T + kc + kk) * DM + h * HD + d]);
    }
    __syncthreads();

    // S = (Q*scale) K^T  (4 key tiles of 16, K-dim 64 = 2 WMMA each)
    v8f s[4];
    for (int tj = 0; tj < 4; ++tj) {
      for (int e = 0; e < 8; ++e) s[tj][e] = 0.f;
      for (int c = 0; c < 2; ++c) {
        v16bf bk;
        const int n = tj * 16 + l15;
        for (int vv = 0; vv < 8; ++vv) {
          const int kk = c * 32 + 2 * vv + 16 * hi;
          bk[2 * vv]     = BF(sK[n][kk]);
          bk[2 * vv + 1] = BF(sK[n][kk + 1]);
        }
        s[tj] = __builtin_amdgcn_wmma_f32_16x16x32_bf16(
            false, aq[c], false, bk, (short)0, s[tj], false, false);
      }
    }

    // bias + online softmax (row r lives at acc element vv=r, lanes of one half)
    float mloc[8];
    for (int r = 0; r < 8; ++r) mloc[r] = -3.0e38f;
    for (int tj = 0; tj < 4; ++tj)
      for (int r = 0; r < 8; ++r) {
        s[tj][r] += qb[r];
        mloc[r] = fmaxf(mloc[r], s[tj][r]);
      }
    for (int off = 1; off < 16; off <<= 1)
      for (int r = 0; r < 8; ++r)
        mloc[r] = fmaxf(mloc[r], __shfl_xor(mloc[r], off, 32));

    float psum[8];
    for (int r = 0; r < 8; ++r) {
      const float mnew = fmaxf(mrow[r], mloc[r]);
      const float sc = __expf(mrow[r] - mnew);
      mrow[r] = mnew;
      lsum[r] *= sc;
      for (int j = 0; j < 4; ++j) o[j][r] *= sc;
      psum[r] = 0.f;
    }
    for (int tj = 0; tj < 4; ++tj)
      for (int r = 0; r < 8; ++r) {
        const float p = __expf(s[tj][r] - mrow[r]);
        psum[r] += p;
        sP[wave][hi * 8 + r][tj * 16 + l15] = f32_to_bf16(p);
      }
    for (int off = 1; off < 16; off <<= 1)
      for (int r = 0; r < 8; ++r) psum[r] += __shfl_xor(psum[r], off, 32);
    for (int r = 0; r < 8; ++r) lsum[r] += psum[r];
    __syncthreads();   // sP visible

    // O += P V
    {
      v16bf ap[2];
      const int m = l15;
      for (int c = 0; c < 2; ++c)
        for (int vv = 0; vv < 8; ++vv) {
          const int kk = c * 32 + 2 * vv + 16 * hi;
          ap[c][2 * vv]     = BF(sP[wave][m][kk]);
          ap[c][2 * vv + 1] = BF(sP[wave][m][kk + 1]);
        }
      for (int tjd = 0; tjd < 4; ++tjd) {
        for (int c = 0; c < 2; ++c) {
          v16bf bv;
          const int n = tjd * 16 + l15;
          for (int vv = 0; vv < 8; ++vv) {
            const int kk = c * 32 + 2 * vv + 16 * hi;
            bv[2 * vv]     = BF(sV[n][kk]);
            bv[2 * vv + 1] = BF(sV[n][kk + 1]);
          }
          o[tjd] = __builtin_amdgcn_wmma_f32_16x16x32_bf16(
              false, ap[c], false, bv, (short)0, o[tjd], false, false);
        }
      }
    }
    __syncthreads();   // before next chunk overwrites sK/sV
  }

  // normalize and write ctx in (B,T,H*HD) layout (ready for @ Wo)
  for (int tjd = 0; tjd < 4; ++tjd) {
    const int d = tjd * 16 + l15;
    for (int r = 0; r < 8; ++r) {
      const int qq = q0 + wave * 16 + hi * 8 + r;
      ctx[((size_t)b * T + qq) * DM + h * HD + d] = o[tjd][r] / lsum[r];
    }
  }
}

// ---------------------------------------------------------------------------
// Host-side orchestration
// ---------------------------------------------------------------------------
extern "C" void kernel_launch(void* const* d_in, const int* in_sizes, int n_in,
                              void* d_out, int out_size, void* d_ws, size_t ws_size,
                              hipStream_t stream) {
  (void)in_sizes; (void)n_in; (void)out_size; (void)ws_size;
  const int B = 4, T = 1024, DM = 1024, H = 16, FF = 4096;
  const int BT = B * T;                 // 4096 rows
  const long NTOT = (long)BT * DM;      // 4M elems

  const float* hs      = (const float*)d_in[0];
  const float* ln1_g   = (const float*)d_in[1];
  const float* ln1_b   = (const float*)d_in[2];
  const float* ln2_g   = (const float*)d_in[3];
  const float* ln2_b   = (const float*)d_in[4];
  const float* Wq      = (const float*)d_in[5];
  const float* Wk      = (const float*)d_in[6];
  const float* Wv      = (const float*)d_in[7];
  const float* Wo      = (const float*)d_in[8];
  const float* bq      = (const float*)d_in[9];
  const float* bk      = (const float*)d_in[10];
  const float* bv      = (const float*)d_in[11];
  const float* bo      = (const float*)d_in[12];
  const float* pe1_w   = (const float*)d_in[13];
  const float* pe1_b   = (const float*)d_in[14];
  const float* pe_ln_g = (const float*)d_in[15];
  const float* pe_ln_b = (const float*)d_in[16];
  const float* pe2_w   = (const float*)d_in[17];
  const float* pe2_b   = (const float*)d_in[18];
  const float* pb_w    = (const float*)d_in[19];
  const float* pb_b    = (const float*)d_in[20];
  const float* f1_w    = (const float*)d_in[21];
  const float* f1_b    = (const float*)d_in[22];
  const float* f2_w    = (const float*)d_in[23];
  const float* f2_b    = (const float*)d_in[24];
  const float* ig1_w   = (const float*)d_in[25];
  const float* ig1_b   = (const float*)d_in[26];
  const float* ig_ln_g = (const float*)d_in[27];
  const float* ig_ln_b = (const float*)d_in[28];
  const float* ig2_w   = (const float*)d_in[29];
  const float* ig2_b   = (const float*)d_in[30];
  const float* r1a_w   = (const float*)d_in[31];
  const float* r1a_b   = (const float*)d_in[32];
  const float* r1_ln_g = (const float*)d_in[33];
  const float* r1_ln_b = (const float*)d_in[34];
  const float* r1b_w   = (const float*)d_in[35];
  const float* r1b_b   = (const float*)d_in[36];
  const float* r2a_w   = (const float*)d_in[37];
  const float* r2a_b   = (const float*)d_in[38];
  const float* r2_ln_g = (const float*)d_in[39];
  const float* r2_ln_b = (const float*)d_in[40];
  const float* r2b_w   = (const float*)d_in[41];
  const float* r2b_b   = (const float*)d_in[42];

  // workspace carve-out (f32)
  size_t off = 0;
  char* wsb = (char*)d_ws;
  auto alloc = [&](size_t elems) -> float* {
    float* p = (float*)(wsb + off);
    off += elems * sizeof(float);
    return p;
  };
  float* x    = alloc((size_t)BT * DM);        // ln1 out; reused as ctx later
  float* hpe  = alloc((size_t)BT * (DM / 2));
  float* phi  = alloc((size_t)BT);
  float* pbias= alloc((size_t)B * H * T);
  float* Qb   = alloc((size_t)BT * DM);
  float* Kb   = alloc((size_t)BT * DM);
  float* Vb   = alloc((size_t)BT * DM);
  float* attn = alloc((size_t)BT * DM);
  float* tmp  = alloc((size_t)BT * DM);        // r1h / r2h
  float* a1   = alloc((size_t)BT);
  float* x1   = alloc((size_t)BT * DM);
  float* x2   = alloc((size_t)BT * DM);
  float* igh  = alloc((size_t)BT * (DM / 4));
  float* gg   = alloc((size_t)BT);
  float* fmid = alloc((size_t)BT * FF);
  float* ffv  = alloc((size_t)BT * DM);
  float* ffn  = alloc((size_t)BT * DM);
  float* a2   = alloc((size_t)BT);
  float* ctx  = x;                              // x dead after QKV/pe1 GEMMs

  const dim3 blk(TPB);
  const int nblend = (int)((NTOT + TPB - 1) / TPB);

  // 1) x = LN1(hs)
  ln_kernel<0><<<BT, blk, 0, stream>>>(hs, ln1_g, ln1_b, x, DM);

  // 2) phi estimator
  gemm_bf16_wmma<0><<<dim3((DM / 2) / 128, BT / 128), blk, 0, stream>>>(
      x, DM, nullptr, pe1_w, pe1_b, hpe, BT, DM / 2, DM);
  ln_kernel<1><<<BT, blk, 0, stream>>>(hpe, pe_ln_g, pe_ln_b, hpe, DM / 2);
  dot_sigmoid_kernel<<<BT, blk, 0, stream>>>(hpe, pe2_w, pe2_b, phi, DM / 2);
  phibias_kernel<<<(B * H * T + TPB - 1) / TPB, blk, 0, stream>>>(
      phi, pb_w, pb_b, pbias, B, H, T);

  // 3) Q, K, V projections
  gemm_bf16_wmma<0><<<dim3(DM / 128, BT / 128), blk, 0, stream>>>(
      x, DM, nullptr, Wq, bq, Qb, BT, DM, DM);
  gemm_bf16_wmma<0><<<dim3(DM / 128, BT / 128), blk, 0, stream>>>(
      x, DM, nullptr, Wk, bk, Kb, BT, DM, DM);
  gemm_bf16_wmma<0><<<dim3(DM / 128, BT / 128), blk, 0, stream>>>(
      x, DM, nullptr, Wv, bv, Vb, BT, DM, DM);

  // 4) attention (flash) -> ctx, then Wo
  attn_flash_kernel<<<dim3(T / 128, B * H), blk, 0, stream>>>(Qb, Kb, Vb, pbias, ctx);
  gemm_bf16_wmma<0><<<dim3(DM / 128, BT / 128), blk, 0, stream>>>(
      ctx, DM, nullptr, Wo, bo, attn, BT, DM, DM);

  // 5) phi-aware residual 1: a1 = head(LN(concat(hs, attn) @ r1a))
  gemm_bf16_wmma<0><<<dim3(DM / 128, BT / 128), blk, 0, stream>>>(
      hs, DM, attn, r1a_w, r1a_b, tmp, BT, DM, 2 * DM);
  ln_kernel<1><<<BT, blk, 0, stream>>>(tmp, r1_ln_g, r1_ln_b, tmp, DM);
  dot_sigmoid_kernel<<<BT, blk, 0, stream>>>(tmp, r1b_w, r1b_b, a1, DM);
  blend_kernel<<<nblend, blk, 0, stream>>>(a1, attn, hs, x1, DM, NTOT);

  // 6) x2 = LN2(x1); integration gate
  ln_kernel<0><<<BT, blk, 0, stream>>>(x1, ln2_g, ln2_b, x2, DM);
  gemm_bf16_wmma<0><<<dim3((DM / 4) / 128, BT / 128), blk, 0, stream>>>(
      x2, DM, nullptr, ig1_w, ig1_b, igh, BT, DM / 4, DM);
  ln_kernel<1><<<BT, blk, 0, stream>>>(igh, ig_ln_g, ig_ln_b, igh, DM / 4);
  dot_sigmoid_kernel<<<BT, blk, 0, stream>>>(igh, ig2_w, ig2_b, gg, DM / 4);

  // 7) FFN: fmid = gelu(x2@f1+b1); f = fmid@f2+b2; ffn = g*f + (1-g)*x2
  gemm_bf16_wmma<1><<<dim3(FF / 128, BT / 128), blk, 0, stream>>>(
      x2, DM, nullptr, f1_w, f1_b, fmid, BT, FF, DM);
  gemm_bf16_wmma<0><<<dim3(DM / 128, BT / 128), blk, 0, stream>>>(
      fmid, FF, nullptr, f2_w, f2_b, ffv, BT, DM, FF);
  blend_kernel<<<nblend, blk, 0, stream>>>(gg, ffv, x2, ffn, DM, NTOT);

  // 8) phi-aware residual 2 -> output
  gemm_bf16_wmma<0><<<dim3(DM / 128, BT / 128), blk, 0, stream>>>(
      x1, DM, ffn, r2a_w, r2a_b, tmp, BT, DM, 2 * DM);
  ln_kernel<1><<<BT, blk, 0, stream>>>(tmp, r2_ln_g, r2_ln_b, tmp, DM);
  dot_sigmoid_kernel<<<BT, blk, 0, stream>>>(tmp, r2b_w, r2b_b, a2, DM);
  blend_kernel<<<nblend, blk, 0, stream>>>(a2, ffn, x1, (float*)d_out, DM, NTOT);
}
